// ValueNetork_61014305407497
// MI455X (gfx1250) — matrix operand order repaired and checked by reading, not compiled, not run on hardware
//
#include <hip/hip_runtime.h>
#include <hip/hip_bf16.h>

// Mamba forward, last-timestep output. MI455X / gfx1250, wave32.
// B=16, L=4096, d_model=d_inner=256, d_state=16, d_conv=4, dt_rank=16.

#define BB   16
#define LL   4096
#define DIN  256
#define DSN  16
#define DTR  16
#define TC   128         // scan chunk length
#define NC   (LL / TC)   // 32 chunks

typedef __attribute__((ext_vector_type(16))) _Float16 v16h;
typedef __attribute__((ext_vector_type(8)))  float    v8f;

// ---------------------------------------------------------------------------
// f32-in / f32-out GEMM via f16 WMMA:  C[M,N] = A[M,K] * W[N,K]^T (+bias[n]).
// Register-blocked: one wave computes 1 M-tile x NTILES N-tiles, so the A
// fragment is loaded once per K-step and fed to NTILES back-to-back WMMAs.
// A-frag (16x32 f16): lane L holds row m=L&15, contiguous K run of 16
//   at offset (L>=16)*16.
// B-frag: lane L holds col n=L&15, contiguous K run of 16 at offset (L>=16)*16.
// D-frag: lane L holds col n=L&15, rows mbase..mbase+7 with mbase=(L>=16)*8.
// ---------------------------------------------------------------------------
template <int K, int NTILES>
__global__ void wmma_gemm_kernel(const float* __restrict__ A,
                                 const float* __restrict__ W,
                                 const float* __restrict__ bias,
                                 float* __restrict__ C,
                                 int M, int N) {
  const int lane = threadIdx.x & 31;
  const int wid  = blockIdx.x * (blockDim.x >> 5) + (threadIdx.x >> 5);
  const int NGRP = (N >> 4) / NTILES;     // N-tile groups per M row
  const int ng   = wid % NGRP;
  const int mt   = wid / NGRP;
  if (mt >= (M >> 4)) return;             // wave-uniform guard (EXEC stays full)

  const int am = (mt << 4) + (lane & 15);
  const int s  = (lane >> 4) << 3;        // 0 or 8
  const int nb = ng * NTILES;             // first N-tile of this wave
  const float* __restrict__ arow = A + (size_t)am * K;

  const float* brow[NTILES];
#pragma unroll
  for (int t = 0; t < NTILES; ++t)
    brow[t] = W + (size_t)(((nb + t) << 4) + (lane & 15)) * K + ((lane >> 4) << 4);

  v8f zero = {};
  v8f acc[NTILES];
#pragma unroll
  for (int t = 0; t < NTILES; ++t) acc[t] = zero;

#pragma unroll
  for (int k0 = 0; k0 < K; k0 += 32) {
    v16h a;
#pragma unroll
    for (int j = 0; j < 8; ++j) {
      a[j]     = (_Float16)arow[k0 + 2 * s + j];
      a[j + 8] = (_Float16)arow[k0 + 2 * s + 8 + j];
    }
#pragma unroll
    for (int t = 0; t < NTILES; ++t) {
      v16h b;
#pragma unroll
      for (int j = 0; j < 16; ++j) b[j] = (_Float16)brow[t][k0 + j];
      acc[t] = __builtin_amdgcn_wmma_f32_16x16x32_f16(false, a, false, b,
                                                      (short)0, acc[t], false, false);
    }
  }

  const int mbase = (mt << 4) + ((lane >> 4) << 3);
#pragma unroll
  for (int t = 0; t < NTILES; ++t) {
    const int bn = ((nb + t) << 4) + (lane & 15);
    const float bv = bias ? bias[bn] : 0.0f;
#pragma unroll
    for (int r = 0; r < 8; ++r)
      C[(size_t)(mbase + r) * N + bn] = acc[t][r] + bv;
  }
}

// ---------------------------------------------------------------------------
// delta = softplus(dt @ W_dt^T + b_dt).  M x 256, K=16 zero-padded to one
// 16x16x32 WMMA per tile.  dt = dtB[:, 0:16] (row stride 32).
// Branch-free softplus: max(v,0) + log1p(exp(-|v|)).
// ---------------------------------------------------------------------------
__global__ void delta_kernel(const float* __restrict__ dtB,
                             const float* __restrict__ W_dt,
                             const float* __restrict__ b_dt,
                             float* __restrict__ delta, int M) {
  const int lane = threadIdx.x & 31;
  const int wid  = blockIdx.x * (blockDim.x >> 5) + (threadIdx.x >> 5);
  const int nt   = wid & 15;
  const int mt   = wid >> 4;
  if (mt >= (M >> 4)) return;

  const int am = (mt << 4) + (lane & 15);
  const int bn = (nt << 4) + (lane & 15);
  const int s  = (lane >> 4) << 3;        // A: K runs {s..s+7} valid, {s+16..} pad
  const float* __restrict__ arow = dtB + (size_t)am * 32;
  const float* __restrict__ brow = W_dt + (size_t)bn * 16;

  v16h a, b;
#pragma unroll
  for (int j = 0; j < 8; ++j) {
    a[j]     = (_Float16)arow[s + j];     // K = s..s+7  (real dt, K<16)
    a[j + 8] = (_Float16)0.0f;            // K = s+16..s+23 (pad)
  }
#pragma unroll
  for (int j = 0; j < 16; ++j)            // K = s..s+7 real, K = s+16..s+23 pad
    b[j] = (j < 8) ? (_Float16)brow[s + j] : (_Float16)0.0f;

  v8f acc = {};
  acc = __builtin_amdgcn_wmma_f32_16x16x32_f16(false, a, false, b,
                                               (short)0, acc, false, false);

  const float bd = b_dt[bn];
  const int mbase = (mt << 4) + ((lane >> 4) << 3);
#pragma unroll
  for (int r = 0; r < 8; ++r) {
    float v  = acc[r] + bd;
    float sp = fmaxf(v, 0.0f) + __logf(1.0f + __expf(-fabsf(v)));
    delta[(size_t)(mbase + r) * DIN + bn] = sp;
  }
}

// ---------------------------------------------------------------------------
// Depthwise causal conv (DC=4) + SiLU.  One block per (b,l), thread = channel.
// ---------------------------------------------------------------------------
__global__ void conv_silu_kernel(const float* __restrict__ xr,
                                 const float* __restrict__ cw,
                                 const float* __restrict__ cb,
                                 float* __restrict__ x) {
  const int d  = threadIdx.x;             // 0..255
  const size_t bl = blockIdx.x;           // b*L + l
  const int l  = (int)(bl & (LL - 1));
  float acc = cb[d];
#pragma unroll
  for (int j = 0; j < 4; ++j) {
    int lj = l - 3 + j;
    if (lj >= 0)
      acc += cw[d * 4 + j] * xr[(bl - (size_t)(3 - j)) * DIN + d];
  }
  x[bl * DIN + d] = acc / (1.0f + __expf(-acc));   // silu
}

// ---------------------------------------------------------------------------
// Scan pass 1: each block handles (b, d-group of 16, chunk of TC timesteps).
// Stages delta/x/B through LDS, runs the 128-step affine recurrence, and
// emits the chunk's composed transform: aP = exp(A * sum(dt)), bP = h_chunk.
// ---------------------------------------------------------------------------
__global__ void scan_chunk_kernel(const float* __restrict__ delta,
                                  const float* __restrict__ x,
                                  const float* __restrict__ dtB,
                                  const float* __restrict__ A_log,
                                  float* __restrict__ aP,
                                  float* __restrict__ bP) {
  __shared__ float sD[TC * 16];
  __shared__ float sX[TC * 16];
  __shared__ float sB[TC * 16];

  const int c   = blockIdx.x & (NC - 1);
  const int dg  = (blockIdx.x >> 5) & 15;
  const int b   = blockIdx.x >> 9;
  const int tid = threadIdx.x;
  const int n   = tid & 15;
  const int dl  = tid >> 4;
  const int d   = (dg << 4) + dl;
  const size_t rowBL = (size_t)b * LL + (size_t)c * TC;

  for (int i = tid; i < TC * 16; i += 256) {
    const int tl = i >> 4, cc = i & 15;
    const size_t r = rowBL + (size_t)tl;
    sD[i] = delta[r * DIN + (dg << 4) + cc];
    sX[i] = x[r * DIN + (dg << 4) + cc];
    sB[i] = dtB[r * 32 + 16 + cc];        // B half of x_proj output
  }
  __syncthreads();

  const float Adn = -__expf(A_log[d * DSN + n]);
  float h = 0.0f, sdt = 0.0f;
#pragma unroll 4
  for (int tl = 0; tl < TC; ++tl) {
    const float dt = sD[(tl << 4) + dl];
    const float xv = sX[(tl << 4) + dl];
    const float bv = sB[(tl << 4) + n];
    h    = __expf(dt * Adn) * h + (dt * bv) * xv;
    sdt += dt;
  }
  const size_t o = (size_t)blockIdx.x * 256 + tid;  // [(b,dg,c), tid]
  aP[o] = __expf(Adn * sdt);              // == prod_t exp(dt*A)
  bP[o] = h;
}

// ---------------------------------------------------------------------------
// Scan pass 2: sequentially compose the NC=32 chunk transforms.
// ---------------------------------------------------------------------------
__global__ void scan_combine_kernel(const float* __restrict__ aP,
                                    const float* __restrict__ bP,
                                    float* __restrict__ hfin) {
  const int tid = threadIdx.x;
  const int bdg = blockIdx.x;             // b*16 + dg, 256 blocks
  float h = 0.0f;
  for (int c = 0; c < NC; ++c) {
    const size_t i = ((size_t)bdg * NC + c) * 256 + tid;
    h = aP[i] * h + bP[i];
  }
  const int b  = bdg >> 4, dg = bdg & 15;
  const int dl = tid >> 4, n = tid & 15;
  hfin[(((size_t)b * DIN) + (dg << 4) + dl) * DSN + n] = h;
}

// ---------------------------------------------------------------------------
// Final: only t = L-1 matters. Compute z_last, C_last, y, gate, W_out dot.
// One block per batch element, thread = channel d.
// ---------------------------------------------------------------------------
__global__ void final_kernel(const float* __restrict__ state,
                             const float* __restrict__ W_in,
                             const float* __restrict__ b_in,
                             const float* __restrict__ x,
                             const float* __restrict__ hfin,
                             const float* __restrict__ W_xproj,
                             const float* __restrict__ Dp,
                             const float* __restrict__ W_out,
                             const float* __restrict__ b_out,
                             float* __restrict__ out) {
  __shared__ float sC[DSN];
  __shared__ float sred[DIN];
  const int b = blockIdx.x, d = threadIdx.x;
  const float* __restrict__ s_last = state + ((size_t)b * LL + (LL - 1)) * DIN;
  const float* __restrict__ x_last = x + ((size_t)b * LL + (LL - 1)) * DIN;

  // z_last[d] = state_last . W_in[DIN + d] + b_in[DIN + d]
  float z = b_in[DIN + d];
  const float* __restrict__ wz = W_in + (size_t)(DIN + d) * DIN;
  for (int k = 0; k < DIN; ++k) z += s_last[k] * wz[k];

  // C_last[n] = x_last . W_xproj[DTR + DSN + n]
  if (d < DSN) {
    const float* __restrict__ wc = W_xproj + (size_t)(DTR + DSN + d) * DIN;
    float cacc = 0.0f;
    for (int k = 0; k < DIN; ++k) cacc += x_last[k] * wc[k];
    sC[d] = cacc;
  }
  __syncthreads();

  const float* __restrict__ hp = hfin + ((size_t)b * DIN + d) * DSN;
  float y = x_last[d] * Dp[d];
#pragma unroll
  for (int n = 0; n < DSN; ++n) y += hp[n] * sC[n];

  const float gate = z / (1.0f + __expf(-z));       // silu(z)
  sred[d] = W_out[d] * (y * gate);
  __syncthreads();
  for (int off = 128; off > 0; off >>= 1) {
    if (d < off) sred[d] += sred[d + off];
    __syncthreads();
  }
  if (d == 0) out[b] = sred[0] + b_out[0];
}

// ---------------------------------------------------------------------------
extern "C" void kernel_launch(void* const* d_in, const int* in_sizes, int n_in,
                              void* d_out, int out_size, void* d_ws, size_t ws_size,
                              hipStream_t stream) {
  (void)in_sizes; (void)n_in; (void)out_size; (void)ws_size;
  const float* state   = (const float*)d_in[0];
  const float* W_in    = (const float*)d_in[1];
  const float* b_in    = (const float*)d_in[2];
  const float* conv_w  = (const float*)d_in[3];
  const float* conv_b  = (const float*)d_in[4];
  const float* W_xproj = (const float*)d_in[5];
  const float* W_dt    = (const float*)d_in[6];
  const float* b_dt    = (const float*)d_in[7];
  const float* A_log   = (const float*)d_in[8];
  const float* Dvec    = (const float*)d_in[9];
  const float* W_out   = (const float*)d_in[10];
  const float* b_out   = (const float*)d_in[11];
  float* out = (float*)d_out;

  const size_t M = (size_t)BB * LL;                 // 65536 rows
  // Workspace layout (~152 MB of f32):
  float* x_raw = (float*)d_ws;                      // [M,256] pre-conv; reused as delta
  float* xbuf  = x_raw + M * DIN;                   // [M,256] post conv+silu
  float* dtB   = xbuf + M * DIN;                    // [M,32]  dt(0:16) | B(16:32)
  float* aP    = dtB + M * 32;                      // [8192,256] chunk transforms
  float* bP    = aP + (size_t)8192 * 256;
  float* hfin  = bP + (size_t)8192 * 256;           // [16,256,16]
  float* delta = x_raw;                             // alias (x_raw dead after conv)

  const int Mi = (int)M;

  // 1) x_raw = state @ W_in[0:256]^T + b_in[0:256]   (8.6 GFLOP, f16 WMMA,
  //    4-way N register blocking: waves = Mtiles * Ntiles/4)
  wmma_gemm_kernel<DIN, 4><<<dim3((Mi / 16) * (DIN / 16) / 4 / 8), dim3(256), 0, stream>>>(
      state, W_in, b_in, x_raw, Mi, DIN);

  // 2) depthwise causal conv + SiLU
  conv_silu_kernel<<<dim3(Mi), dim3(256), 0, stream>>>(x_raw, conv_w, conv_b, xbuf);

  // 3) dtB = x @ W_xproj[0:32]^T   (dt_rank + d_state columns, no bias,
  //    2-way N blocking: one wave covers the full N=32)
  wmma_gemm_kernel<DIN, 2><<<dim3((Mi / 16) * (32 / 16) / 2 / 8), dim3(256), 0, stream>>>(
      xbuf, W_xproj, nullptr, dtB, Mi, 32);

  // 4) delta = softplus(dt @ W_dt^T + b_dt)   (K=16 padded WMMA; overwrites x_raw)
  delta_kernel<<<dim3((Mi / 16) * (DIN / 16) / 8), dim3(256), 0, stream>>>(
      dtB, W_dt, b_dt, delta, Mi);

  // 5) chunked selective scan, pass 1: 16 b x 16 d-groups x 32 chunks
  scan_chunk_kernel<<<dim3(BB * 16 * NC), dim3(256), 0, stream>>>(
      delta, xbuf, dtB, A_log, aP, bP);

  // 6) compose chunk transforms -> h at t = L-1
  scan_combine_kernel<<<dim3(BB * 16), dim3(256), 0, stream>>>(aP, bP, hfin);

  // 7) last-timestep z, C, gate, output projection -> out[16]
  final_kernel<<<dim3(BB), dim3(256), 0, stream>>>(
      state, W_in, b_in, xbuf, hfin, W_xproj, Dvec, W_out, b_out, out);
}
